// GemmaAttention_28518582845841
// MI455X (gfx1250) — matrix operand order repaired and checked by reading, not compiled, not run on hardware
//
#include <hip/hip_runtime.h>
#include <hip/hip_bf16.h>
#include <stdint.h>

// ---------------- problem constants (match reference) ----------------
constexpr int Bc = 2;
constexpr int Sc = 2048;
constexpr int Dc = 2048;
constexpr int Hc = 16;
constexpr int HDc = 128;          // head dim
constexpr int Mc = Bc * Sc;       // 4096 rows
constexpr float SCALE = 0.08838834764831845f;   // HD^-0.5
constexpr float LN_ROPE_BASE = 9.210340371976184f; // ln(10000)

// ---------------- WMMA vector types ----------------
typedef __attribute__((ext_vector_type(16))) __bf16 v16bf;
typedef __attribute__((ext_vector_type(8)))  __bf16 bf16x8;
typedef __attribute__((ext_vector_type(8)))  float  v8f;

// ---------------- TDM descriptor vector types ----------------
typedef __attribute__((ext_vector_type(4))) unsigned int u32x4;
typedef __attribute__((ext_vector_type(4))) int          i32x4;
typedef __attribute__((ext_vector_type(8))) int          i32x8;

#if __has_builtin(__builtin_amdgcn_tensor_load_to_lds) && \
    __has_builtin(__builtin_amdgcn_s_wait_tensorcnt)
#define HAVE_TDM 1
#else
#define HAVE_TDM 0
#endif

__device__ __forceinline__ v8f wmma_bf16(v16bf a, v16bf b, v8f c) {
  // emits v_wmma_f32_16x16x32_bf16
  return __builtin_amdgcn_wmma_f32_16x16x32_bf16(
      /*neg_a=*/false, a, /*neg_b=*/false, b,
      /*c_mod=*/(short)0, c, /*reuse_a=*/false, /*reuse_b=*/false);
}

__device__ __forceinline__ v8f zero8() {
  v8f z = {0.f, 0.f, 0.f, 0.f, 0.f, 0.f, 0.f, 0.f};
  return z;
}

// ----- A-fragment (16x32 bf16, MxK) from fp32 row-major [row, k] -----
__device__ __forceinline__ v16bf load_a_f32(const float* __restrict__ base,
                                            int ld, int row, int k0, int lane) {
  const int kh = (lane >= 16) ? 16 : 0;
  const float* p = base + (size_t)row * ld + k0 + kh;
  float4 a0 = *(const float4*)(p);
  float4 a1 = *(const float4*)(p + 4);
  float4 b0 = *(const float4*)(p + 8);
  float4 b1 = *(const float4*)(p + 12);
  v16bf r;
  r[0]=(__bf16)a0.x;  r[1]=(__bf16)a0.y;  r[2]=(__bf16)a0.z;  r[3]=(__bf16)a0.w;
  r[4]=(__bf16)a1.x;  r[5]=(__bf16)a1.y;  r[6]=(__bf16)a1.z;  r[7]=(__bf16)a1.w;
  r[8]=(__bf16)b0.x;  r[9]=(__bf16)b0.y;  r[10]=(__bf16)b0.z; r[11]=(__bf16)b0.w;
  r[12]=(__bf16)b1.x; r[13]=(__bf16)b1.y; r[14]=(__bf16)b1.z; r[15]=(__bf16)b1.w;
  return r;
}

// ----- B-fragment (32x16 bf16, KxN) where B(k,n) = Wrow_n[k] (fp32) -----
__device__ __forceinline__ v16bf load_b_f32(const float* __restrict__ base,
                                            int ld, int nrow, int k0, int lane) {
  const int kh = (lane >= 16) ? 16 : 0;
  const float* p = base + (size_t)nrow * ld + k0 + kh;
  float4 a0 = *(const float4*)(p);
  float4 a1 = *(const float4*)(p + 4);
  float4 a2 = *(const float4*)(p + 8);
  float4 a3 = *(const float4*)(p + 12);
  v16bf r;
  r[0]=(__bf16)a0.x;  r[1]=(__bf16)a0.y;  r[2]=(__bf16)a0.z;  r[3]=(__bf16)a0.w;
  r[4]=(__bf16)a1.x;  r[5]=(__bf16)a1.y;  r[6]=(__bf16)a1.z;  r[7]=(__bf16)a1.w;
  r[8]=(__bf16)a2.x;  r[9]=(__bf16)a2.y;  r[10]=(__bf16)a2.z; r[11]=(__bf16)a2.w;
  r[12]=(__bf16)a3.x; r[13]=(__bf16)a3.y; r[14]=(__bf16)a3.z; r[15]=(__bf16)a3.w;
  return r;
}

__device__ __forceinline__ v16bf frag_from2(bf16x8 lo, bf16x8 hi) {
  v16bf r;
#pragma unroll
  for (int i = 0; i < 8; ++i) { r[i] = lo[i]; r[8 + i] = hi[i]; }
  return r;
}

__device__ __forceinline__ v16bf load_a_bf16(const __bf16* base, int ld,
                                             int row, int k0, int lane) {
  const int kh = (lane >= 16) ? 16 : 0;
  const __bf16* p = base + (size_t)row * ld + k0 + kh;
  return frag_from2(*(const bf16x8*)(p), *(const bf16x8*)(p + 8));
}

__device__ __forceinline__ v16bf load_frag16(const __bf16* p) {
  return frag_from2(*(const bf16x8*)(p), *(const bf16x8*)(p + 8));
}

// ==================================================================
// Kernel 1: fused QKV projection (bf16 WMMA, f32 accum) + RoPE epilogue.
// ==================================================================
__global__ __launch_bounds__(128) void qkv_rope_kernel(
    const float* __restrict__ X,
    const float* __restrict__ Wq, const float* __restrict__ Wk,
    const float* __restrict__ Wv,
    __bf16* __restrict__ Qr, __bf16* __restrict__ Kr, __bf16* __restrict__ Vb) {
  const int lane = threadIdx.x & 31;
  const int wave = threadIdx.x >> 5;
  const int la = lane & 15;
  const int half8 = (lane >= 16) ? 8 : 0;

  const int m_base = blockIdx.x * 64 + wave * 16;
  const int h = blockIdx.y >> 2;
  const int idx0 = (blockIdx.y & 3) * 16;   // rotary pair-index base (0..48)
  const int z = blockIdx.z;

  const float* W = (z == 0) ? Wq : (z == 1) ? Wk : Wv;
  __bf16* Out = (z == 0) ? Qr : (z == 1) ? Kr : Vb;

  const int n1 = h * HDc + idx0 + la;  // W row for tile 1 (B(k,n)=W[n][k])
  const int n2 = n1 + 64;              // rotate_half partner column
  const int arow = m_base + la;

  v8f acc1 = zero8(), acc2 = zero8();

  for (int k0 = 0; k0 < Dc; k0 += 32) {
    if (k0 + 64 < Dc) {   // next-tile prefetch -> global_prefetch_b8
      __builtin_prefetch(X + (size_t)arow * Dc + k0 + 64, 0, 1);
      __builtin_prefetch(W + (size_t)n1 * Dc + k0 + 64, 0, 1);
      __builtin_prefetch(W + (size_t)n2 * Dc + k0 + 64, 0, 1);
    }
    v16bf a  = load_a_f32(X, Dc, arow, k0, lane);
    v16bf b1 = load_b_f32(W, Dc, n1, k0, lane);
    v16bf b2 = load_b_f32(W, Dc, n2, k0, lane);
    acc1 = wmma_bf16(a, b1, acc1);
    acc2 = wmma_bf16(a, b2, acc2);
  }

  const int idx = idx0 + la;
  const float invf = __expf((float)idx * (-2.0f / (float)HDc) * LN_ROPE_BASE);
#pragma unroll
  for (int e = 0; e < 8; ++e) {
    const int m = m_base + e + half8;
    const int b = m >> 11;       // / S
    const int s = m & (Sc - 1);  // % S
    const size_t o1 = ((size_t)(b * Hc + h) * Sc + s) * HDc + idx;
    const float v1 = acc1[e], v2 = acc2[e];
    if (z < 2) {
      float sn, cs;
      __sincosf((float)s * invf, &sn, &cs);
      Out[o1]      = (__bf16)(v1 * cs - v2 * sn);
      Out[o1 + 64] = (__bf16)(v2 * cs + v1 * sn);
    } else {
      Out[o1]      = (__bf16)v1;
      Out[o1 + 64] = (__bf16)v2;
    }
  }
}

// ==================================================================
// Kernel 2: flash attention with TDM-staged K tiles.
// grid = (B*H, S/64), block = 128 (4 waves).
// Per 32-key step: K tile (32x128 bf16, 8KB contiguous) DMA'd to LDS by
// the Tensor Data Mover (wave 0 issues, drains with s_wait_tensorcnt),
// overlapped with cooperative transposed V staging by all threads.
// ==================================================================
#define VPAD 40   // padded key-stride for transposed V (keeps 16B alignment)

__global__ __launch_bounds__(128) void flash_attn_kernel(
    const __bf16* __restrict__ Qr, const __bf16* __restrict__ Kr,
    const __bf16* __restrict__ Vb, const float* __restrict__ mask,
    __bf16* __restrict__ ctx) {
  __shared__ __bf16 Ks[32][HDc];       // 8 KB: keys row-major (TDM target)
  __shared__ __bf16 Vst[HDc][VPAD];    // 10 KB: V transposed [hd][key]
  __shared__ __bf16 Pw[4][16][32];     // 4 KB: per-wave P scratch

  const int tid = threadIdx.x;
  const int lane = tid & 31;
  const int wave = tid >> 5;
  const int la = lane & 15;
  const int half8 = (lane >= 16) ? 8 : 0;
  const int kh16 = (lane >= 16) ? 16 : 0;

  const int bh = blockIdx.x;          // 0..31
  const int b = bh >> 4, h = bh & 15;
  const int q0 = blockIdx.y * 64 + wave * 16;

  const __bf16* Qh = Qr + (size_t)bh * Sc * HDc;
  const __bf16* Kh = Kr + (size_t)bh * Sc * HDc;
  const __bf16* Vh = Vb + (size_t)bh * Sc * HDc;
  const float* mk = mask + (size_t)b * Sc * Sc;

  // Q fragments for the whole 16x128 strip, held in registers.
  v16bf qf[4];
#pragma unroll
  for (int kk = 0; kk < 4; ++kk)
    qf[kk] = load_a_bf16(Qh, HDc, q0 + la, kk * 32, lane);

  v8f o[8];
#pragma unroll
  for (int t = 0; t < 8; ++t) o[t] = zero8();
  float mrow[8], lrow[8];
#pragma unroll
  for (int e = 0; e < 8; ++e) { mrow[e] = -1e30f; lrow[e] = 0.f; }

#if HAVE_TDM
  // LDS byte offset of Ks: low 32 bits of the generic shared address.
  const uint32_t ks_lds = (uint32_t)(uintptr_t)&Ks[0][0];
#endif

  for (int j = 0; j < Sc; j += 32) {
#if HAVE_TDM
    // ---- K tile via Tensor Data Mover (wave 0 only, EXEC-independent) ----
    if (wave == 0) {
      const uint64_t ga = (uint64_t)(uintptr_t)(Kh + (size_t)j * HDc);
      // D# group 0: count=1 | lds_addr | global_addr[56:0] | type=2
      u32x4 g0;
      g0[0] = 1u;                                     // count=1 (valid user D#)
      g0[1] = ks_lds;                                 // lds_addr (bytes)
      g0[2] = (uint32_t)ga;                           // global_addr[31:0]
      g0[3] = (uint32_t)((ga >> 32) & 0x01FFFFFFu) | (2u << 30); // [56:32]|type
      // D# group 1: data_size=2B; tensor 128x2048; tile 128x32; stride0=128
      i32x8 g1;
      g1[0] = (int)(1u << 16);                        // data_size=1 -> 2 bytes
      g1[1] = (int)((128u & 0xFFFFu) << 16);          // tensor_dim0 lo16
      g1[2] = (int)((128u >> 16) | ((2048u & 0xFFFFu) << 16)); // d0 hi|d1 lo
      g1[3] = (int)((2048u >> 16) | (128u << 16));    // d1 hi | tile_dim0=128
      g1[4] = (int)32u;                               // tile_dim1=32, tile_dim2=0
      g1[5] = (int)128u;                              // tensor_dim0_stride lo32
      g1[6] = 0;                                      // stride0 hi | stride1 lo
      g1[7] = 0;                                      // stride1 hi
      const i32x4 gz = {0, 0, 0, 0};                  // groups 2/3: 2-D tile
#if defined(__clang_major__) && __clang_major__ >= 23
      const i32x8 gz8 = {0, 0, 0, 0, 0, 0, 0, 0};
      __builtin_amdgcn_tensor_load_to_lds(g0, g1, gz, gz, gz8, 0);
#else
      __builtin_amdgcn_tensor_load_to_lds(g0, g1, gz, gz, 0);
#endif
    }
    // ---- overlapped cooperative staging: V transposed ----
#pragma unroll
    for (int i = 0; i < 4; ++i) {
      const int c = i * 128 + tid;       // 0..511 chunk id
      const int r = c >> 4;              // key row in block 0..31
      const int col = (c & 15) * 8;      // hd column
      bf16x8 vv = *(const bf16x8*)(Vh + (size_t)(j + r) * HDc + col);
#pragma unroll
      for (int x = 0; x < 8; ++x) Vst[col + x][r] = vv[x];
    }
    if (wave == 0) __builtin_amdgcn_s_wait_tensorcnt(0);  // drain TDM
    __syncthreads();
#else
    // ---- fallback: manual staging of both K (row-major) and V (transposed)
#pragma unroll
    for (int i = 0; i < 4; ++i) {
      const int c = i * 128 + tid;
      const int r = c >> 4;
      const int col = (c & 15) * 8;
      bf16x8 kv = *(const bf16x8*)(Kh + (size_t)(j + r) * HDc + col);
      *(bf16x8*)(&Ks[r][col]) = kv;
      bf16x8 vv = *(const bf16x8*)(Vh + (size_t)(j + r) * HDc + col);
#pragma unroll
      for (int x = 0; x < 8; ++x) Vst[col + x][r] = vv[x];
    }
    __syncthreads();
#endif

    // ---- scores: two 16(M q-rows) x 16(N keys) tiles, K=128 in 4 chunks
    v8f sa = zero8(), sb = zero8();
#pragma unroll
    for (int kk = 0; kk < 4; ++kk) {
      v16bf ba = load_frag16(&Ks[la][kk * 32 + kh16]);        // keys j..j+15
      v16bf bb = load_frag16(&Ks[16 + la][kk * 32 + kh16]);   // keys j+16..j+31
      sa = wmma_bf16(qf[kk], ba, sa);
      sb = wmma_bf16(qf[kk], bb, sb);
    }

    // ---- online softmax (row = e + half8, cols striped over 16 lanes) ----
#pragma unroll
    for (int e = 0; e < 8; ++e) {
      const int q = q0 + e + half8;
      const float ma = mk[(size_t)q * Sc + j + la];
      const float mb = mk[(size_t)q * Sc + j + 16 + la];
      float va = sa[e] * SCALE + ma;
      float vb_ = sb[e] * SCALE + mb;
      float rm = fmaxf(va, vb_);
#pragma unroll
      for (int d2 = 1; d2 < 16; d2 <<= 1) rm = fmaxf(rm, __shfl_xor(rm, d2, 32));
      const float mn = fmaxf(mrow[e], rm);
      const float corr = __expf(mrow[e] - mn);
      const float pa = __expf(va - mn);
      const float pb = __expf(vb_ - mn);
      float rs = pa + pb;
#pragma unroll
      for (int d2 = 1; d2 < 16; d2 <<= 1) rs += __shfl_xor(rs, d2, 32);
      lrow[e] = lrow[e] * corr + rs;
      mrow[e] = mn;
#pragma unroll
      for (int t = 0; t < 8; ++t) o[t][e] *= corr;
      Pw[wave][e + half8][la] = (__bf16)pa;
      Pw[wave][e + half8][16 + la] = (__bf16)pb;
    }

    // ---- O += P(16x32) @ V(32x128) ----
    v16bf pf = load_a_bf16(&Pw[wave][0][0], 32, la, 0, lane);
#pragma unroll
    for (int t = 0; t < 8; ++t) {
      v16bf vf = load_frag16(&Vst[t * 16 + la][kh16]);  // K=keys contiguous
      o[t] = wmma_bf16(pf, vf, o[t]);
    }
    __syncthreads();
  }

  // ---- epilogue: normalize, store bf16 ctx in [B,S,D] (d = h*HD+hd) ----
#pragma unroll
  for (int e = 0; e < 8; ++e) {
    const float rl = 1.0f / lrow[e];
    const int q = q0 + e + half8;
    __bf16* dst = ctx + (size_t)(b * Sc + q) * Dc + h * HDc;
#pragma unroll
    for (int t = 0; t < 8; ++t) dst[t * 16 + la] = (__bf16)(o[t][e] * rl);
  }
}

// ==================================================================
// Kernel 3: out = ctx @ Wo^T (fp32 result). grid = (M/64, D/16), block 128.
// ==================================================================
__global__ __launch_bounds__(128) void out_proj_kernel(
    const __bf16* __restrict__ ctx, const float* __restrict__ Wo,
    float* __restrict__ out) {
  const int lane = threadIdx.x & 31;
  const int wave = threadIdx.x >> 5;
  const int la = lane & 15;
  const int half8 = (lane >= 16) ? 8 : 0;

  const int m_base = blockIdx.x * 64 + wave * 16;
  const int n = blockIdx.y * 16 + la;   // per-lane output column / Wo row

  v8f acc = zero8();
  for (int k0 = 0; k0 < Dc; k0 += 32) {
    if (k0 + 64 < Dc) {
      __builtin_prefetch(ctx + (size_t)(m_base + la) * Dc + k0 + 64, 0, 1);
      __builtin_prefetch(Wo + (size_t)n * Dc + k0 + 64, 0, 1);
    }
    v16bf a = load_a_bf16(ctx, Dc, m_base + la, k0, lane);
    v16bf bfr = load_b_f32(Wo, Dc, n, k0, lane);
    acc = wmma_bf16(a, bfr, acc);
  }
#pragma unroll
  for (int e = 0; e < 8; ++e)
    out[(size_t)(m_base + e + half8) * Dc + n] = acc[e];
}

// ==================================================================
extern "C" void kernel_launch(void* const* d_in, const int* in_sizes, int n_in,
                              void* d_out, int out_size, void* d_ws,
                              size_t ws_size, hipStream_t stream) {
  (void)in_sizes; (void)n_in; (void)out_size; (void)ws_size;
  const float* X    = (const float*)d_in[0];  // [B,S,D]
  const float* mask = (const float*)d_in[1];  // [B,1,S,S]
  const float* Wq   = (const float*)d_in[2];
  const float* Wk   = (const float*)d_in[3];
  const float* Wv   = (const float*)d_in[4];
  const float* Wo   = (const float*)d_in[5];
  float* out = (float*)d_out;

  const size_t nElem = (size_t)Bc * Hc * Sc * HDc;  // 8M bf16 elems each
  __bf16* Qr  = (__bf16*)d_ws;
  __bf16* Kr  = Qr + nElem;
  __bf16* Vb  = Kr + nElem;
  __bf16* ctx = Vb + nElem;   // 64 MB total workspace

  qkv_rope_kernel<<<dim3(Mc / 64, 64, 3), 128, 0, stream>>>(
      X, Wq, Wk, Wv, Qr, Kr, Vb);
  flash_attn_kernel<<<dim3(Bc * Hc, Sc / 64), 128, 0, stream>>>(
      Qr, Kr, Vb, mask, ctx);
  out_proj_kernel<<<dim3(Mc / 64, Dc / 16), 128, 0, stream>>>(ctx, Wo, out);
}
